// ThreeStageSGNN_65738769432888
// MI455X (gfx1250) — matrix operand order repaired and checked by reading, not compiled, not run on hardware
//
#include <hip/hip_runtime.h>
#include <hip/hip_bf16.h>
#include <math.h>

// ---------------- problem constants (from reference) ----------------
#define N_NODES 100000
#define N_EDGES 1600000
#define N_PRED  100000
#define IN_CH   128
#define HID     64
#define EDGE_THRESHOLD 1e-3f

// ---------------- WMMA types (CDNA5 / gfx1250, wave32) ----------------
typedef __attribute__((ext_vector_type(16))) __bf16 v16bf;
typedef __attribute__((ext_vector_type(8)))  float  v8f;

struct alignas(16) U128 { unsigned int x, y, z, w; };
union Frag { U128 u[2]; v16bf v; };

__device__ __forceinline__ unsigned short f2bf(float f) {
  unsigned int u = __float_as_uint(f);
  u += 0x7FFFu + ((u >> 16) & 1u);          // round-to-nearest-even
  return (unsigned short)(u >> 16);
}
__device__ __forceinline__ float bf2f(unsigned short h) {
  return __uint_as_float(((unsigned int)h) << 16);
}

// ---------------- small element-wise kernels ----------------
__global__ void k_f32_to_bf16(const float* __restrict__ src,
                              unsigned short* __restrict__ dst, int n) {
  int i = blockIdx.x * blockDim.x + threadIdx.x;
  if (i < n) dst[i] = f2bf(src[i]);
}

// src[rows][cols] f32 -> dst[cols][rows] bf16 (so B-columns are contiguous K)
__global__ void k_transpose_bf16(const float* __restrict__ src,
                                 unsigned short* __restrict__ dst,
                                 int rows, int cols) {
  int i = blockIdx.x * blockDim.x + threadIdx.x;
  if (i < rows * cols) {
    int r = i / cols, c = i % cols;
    dst[(size_t)c * rows + r] = f2bf(src[i]);
  }
}

__global__ void k_zero(float* p) { p[0] = 0.0f; }

// ---------------- dual-B WMMA GEMM ----------------
// C0 = A @ B0, C1 = A @ B1. A: [nrows, K] bf16 row-major.
// Bt0/Bt1: B transposed, [64][K] bf16 (column of B contiguous in K).
// Block = 128 threads = 4 waves; wave w handles output columns [16w,16w+16);
// block handles 16 rows. A fragments are loaded once and feed both WMMAs.
__global__ void k_dual_gemm_wmma(const unsigned short* __restrict__ A,
                                 const unsigned short* __restrict__ Bt0,
                                 const unsigned short* __restrict__ Bt1,
                                 float* __restrict__ C0,
                                 float* __restrict__ C1,
                                 int nrows, int K) {
  const int lane = threadIdx.x & 31;
  const int wave = threadIdx.x >> 5;   // 0..3 -> 16-col tile
  const int row0 = blockIdx.x * 16;
  const int n0   = wave * 16;

  int rA = row0 + (lane & 15);
  if (rA >= nrows) rA = nrows - 1;     // clamp loads, guard stores
  const int kbA = (lane >> 4) * 8;     // A layout: 8 contig K at kbA, 8 at kbA+16
  const int col = n0 + (lane & 15);
  const int kbB = (lane >> 4) * 8;     // B layout: 8 contig K at kbB, 8 at kbB+16

  const unsigned short* aRow  = A   + (size_t)rA  * K;
  const unsigned short* b0Row = Bt0 + (size_t)col * K;
  const unsigned short* b1Row = Bt1 + (size_t)col * K;

  v8f c0 = {}; v8f c1 = {};
  for (int kt = 0; kt < K; kt += 32) {
    Frag a, b0, b1;
    a.u[0]  = *(const U128*)(aRow  + kt + kbA);
    a.u[1]  = *(const U128*)(aRow  + kt + kbA + 16);
    b0.u[0] = *(const U128*)(b0Row + kt + kbB);
    b0.u[1] = *(const U128*)(b0Row + kt + kbB + 16);
    b1.u[0] = *(const U128*)(b1Row + kt + kbB);
    b1.u[1] = *(const U128*)(b1Row + kt + kbB + 16);
    c0 = __builtin_amdgcn_wmma_f32_16x16x32_bf16(false, a.v, false, b0.v,
                                                 (short)0, c0, false, false);
    c1 = __builtin_amdgcn_wmma_f32_16x16x32_bf16(false, a.v, false, b1.v,
                                                 (short)0, c1, false, false);
  }

  const int mBase = (lane >> 4) * 8;   // C: M = vgpr + 8*(lane>=16), N = lane&15
  const int nn    = lane & 15;
  for (int r = 0; r < 8; ++r) {
    int row = row0 + mBase + r;
    if (row < nrows) {
      size_t o = (size_t)row * HID + n0 + nn;
      C0[o] = c0[r];
      C1[o] = c1[r];
    }
  }
}

// ---------------- edge gather/scale/scatter-add ----------------
// ACC[dst][c] += XN[src][c] * w   (ACC already holds the x@W_self term)
__global__ void k_edge_aggregate(const float* __restrict__ XN,
                                 const int* __restrict__ src,
                                 const int* __restrict__ dst,
                                 const float* __restrict__ w,
                                 float* __restrict__ ACC,
                                 long long total) {
  long long gid = (long long)blockIdx.x * blockDim.x + threadIdx.x;
  if (gid >= total) return;
  int e = (int)(gid >> 6);
  int c = (int)(gid & 63);
  float we = w[e];
  if (we == 0.0f) return;
  int s = src[e], d = dst[e];
  atomicAdd(&ACC[(size_t)d * HID + c], XN[(size_t)s * HID + c] * we);
}

// H = tanh(S + bias) stored as bf16
__global__ void k_act_bf16(const float* __restrict__ S,
                           const float* __restrict__ bias,
                           unsigned short* __restrict__ Hb, int nrows) {
  int i = blockIdx.x * blockDim.x + threadIdx.x;
  if (i < nrows * HID) {
    int c = i & 63;
    Hb[i] = f2bf(tanhf(S[i] + bias[c]));
  }
}

// per-node dot products with w_src / w_dst
__global__ void k_node_scores(const unsigned short* __restrict__ Hb,
                              const float* __restrict__ wsrc,
                              const float* __restrict__ wdst,
                              float* __restrict__ ssrc,
                              float* __restrict__ sdst, int n) {
  int i = blockIdx.x * blockDim.x + threadIdx.x;
  if (i >= n) return;
  const unsigned short* h = Hb + (size_t)i * HID;
  float a = 0.0f, b = 0.0f;
  for (int c = 0; c < HID; ++c) {
    float v = bf2f(h[c]);
    a += v * wsrc[c];
    b += v * wdst[c];
  }
  ssrc[i] = a;
  sdst[i] = b;
}

// gumbel-sigmoid probs, refined weights, partial sum of probs for l1_reg
__global__ void k_edge_probs(const int* __restrict__ src,
                             const int* __restrict__ dst,
                             const float* __restrict__ w,
                             const float* __restrict__ gumbel,
                             const float* __restrict__ ssrc,
                             const float* __restrict__ sdst,
                             const float* __restrict__ be,
                             float* __restrict__ rw,
                             float* __restrict__ acc, int E) {
  __shared__ float red[256];
  int i = blockIdx.x * blockDim.x + threadIdx.x;
  float p = 0.0f;
  if (i < E) {
    float z = ssrc[src[i]] + sdst[dst[i]] + be[0] + gumbel[i];  // TEMP == 1
    p = 1.0f / (1.0f + expf(-z));
    float r = w[i] * p;
    rw[i] = (fabsf(r) > EDGE_THRESHOLD) ? r : 0.0f;
  }
  red[threadIdx.x] = p;
  __syncthreads();
  for (int s2 = 128; s2 > 0; s2 >>= 1) {
    if ((int)threadIdx.x < s2) red[threadIdx.x] += red[threadIdx.x + s2];
    __syncthreads();
  }
  if (threadIdx.x == 0) atomicAdd(acc, red[0]);
}

// ---------------- predictor GEMM with gathered A ----------------
// hidden = relu( concat(h1[s],h2[s],h1[d],h2[d]) @ Wp1 + b_p1 )
// A row (K=256) assembled per 8-wide K chunk: chunk never crosses a 64-segment.
__global__ void k_pred_gemm_wmma(const unsigned short* __restrict__ h1b,
                                 const unsigned short* __restrict__ h2b,
                                 const int* __restrict__ psrc,
                                 const int* __restrict__ pdst,
                                 const unsigned short* __restrict__ Wp1t, // [64][256]
                                 const float* __restrict__ bp1,
                                 float* __restrict__ Hidden, int P) {
  const int lane = threadIdx.x & 31;
  const int wave = threadIdx.x >> 5;
  const int row0 = blockIdx.x * 16;
  const int n0   = wave * 16;

  int rA = row0 + (lane & 15);
  if (rA >= P) rA = P - 1;
  const int s = psrc[rA], d = pdst[rA];
  const int kbA = (lane >> 4) * 8;
  const int col = n0 + (lane & 15);
  const int kbB = (lane >> 4) * 8;
  const unsigned short* bRow = Wp1t + (size_t)col * (4 * HID);

  v8f c = {};
  for (int kt = 0; kt < 4 * HID; kt += 32) {
    Frag a, b;
    int k0 = kt + kbA;
    int k1 = k0 + 16;
    const unsigned short* p0 =
        ((k0 & 64) ? h2b : h1b) + (size_t)((k0 & 128) ? d : s) * HID + (k0 & 63);
    const unsigned short* p1 =
        ((k1 & 64) ? h2b : h1b) + (size_t)((k1 & 128) ? d : s) * HID + (k1 & 63);
    a.u[0] = *(const U128*)p0;
    a.u[1] = *(const U128*)p1;
    b.u[0] = *(const U128*)(bRow + kt + kbB);
    b.u[1] = *(const U128*)(bRow + kt + kbB + 16);
    c = __builtin_amdgcn_wmma_f32_16x16x32_bf16(false, a.v, false, b.v,
                                                (short)0, c, false, false);
  }

  const int mBase = (lane >> 4) * 8;
  const int nn    = lane & 15;
  float bb = bp1[n0 + nn];
  for (int r = 0; r < 8; ++r) {
    int row = row0 + mBase + r;
    if (row < P) {
      float v = c[r] + bb;
      Hidden[(size_t)row * HID + n0 + nn] = v > 0.0f ? v : 0.0f;
    }
  }
}

// logits = hidden @ W_p2 + b_p2   (K=64, 2 classes -> plain FMA)
__global__ void k_final_logits(const float* __restrict__ Hidden,
                               const float* __restrict__ Wp2,
                               const float* __restrict__ bp2,
                               float* __restrict__ out, int P) {
  int i = blockIdx.x * blockDim.x + threadIdx.x;
  if (i >= P) return;
  const float* h = Hidden + (size_t)i * HID;
  float a0 = bp2[0], a1 = bp2[1];
  for (int k = 0; k < HID; ++k) {
    float v = h[k];
    a0 += v * Wp2[k * 2 + 0];
    a1 += v * Wp2[k * 2 + 1];
  }
  out[(size_t)i * 2 + 0] = a0;
  out[(size_t)i * 2 + 1] = a1;
}

__global__ void k_write_l1(const float* __restrict__ acc, float* __restrict__ out,
                           int E, int P) {
  out[(size_t)2 * P] = acc[0] / (float)E;
}

// ---------------- host launcher ----------------
extern "C" void kernel_launch(void* const* d_in, const int* in_sizes, int n_in,
                              void* d_out, int out_size, void* d_ws, size_t ws_size,
                              hipStream_t stream) {
  (void)in_sizes; (void)n_in; (void)out_size; (void)ws_size;

  const float* x    = (const float*)d_in[0];
  const int*   ei   = (const int*)d_in[1];
  const float* ew   = (const float*)d_in[2];
  const int*   pei  = (const int*)d_in[3];
  const float* gum  = (const float*)d_in[4];
  const float* Ws1  = (const float*)d_in[5];
  const float* Wn1  = (const float*)d_in[6];
  const float* b1   = (const float*)d_in[7];
  const float* wsrc = (const float*)d_in[8];
  const float* wdst = (const float*)d_in[9];
  const float* be   = (const float*)d_in[10];
  const float* Ws2  = (const float*)d_in[11];
  const float* Wn2  = (const float*)d_in[12];
  const float* b2   = (const float*)d_in[13];
  const float* Wp1  = (const float*)d_in[14];
  const float* bp1  = (const float*)d_in[15];
  const float* Wp2  = (const float*)d_in[16];
  const float* bp2  = (const float*)d_in[17];
  float* out = (float*)d_out;

  const int* src  = ei;
  const int* dst  = ei + N_EDGES;
  const int* psrc = pei;
  const int* pdst = pei + N_PRED;

  // workspace layout
  char* ws = (char*)d_ws;
  size_t off = 0;
  auto alloc = [&](size_t bytes) -> void* {
    void* p = ws + off;
    off = (off + bytes + 255) & ~(size_t)255;
    return p;
  };
  unsigned short* xb   = (unsigned short*)alloc((size_t)N_NODES * IN_CH * 2); // x bf16
  unsigned short* wts1 = (unsigned short*)alloc((size_t)HID * IN_CH * 2);     // W_self1^T
  unsigned short* wtn1 = (unsigned short*)alloc((size_t)HID * IN_CH * 2);     // W_neigh1^T
  unsigned short* wts2 = (unsigned short*)alloc((size_t)HID * HID * 2);       // W_self2^T
  unsigned short* wtn2 = (unsigned short*)alloc((size_t)HID * HID * 2);       // W_neigh2^T
  unsigned short* wtp1 = (unsigned short*)alloc((size_t)HID * 4 * HID * 2);   // W_p1^T
  float* XS  = (float*)alloc((size_t)N_NODES * HID * 4); // self term (+agg), reused stage3
  float* XN  = (float*)alloc((size_t)N_NODES * HID * 4); // neigh term, reused as hidden
  unsigned short* h1b = (unsigned short*)alloc((size_t)N_NODES * HID * 2);
  unsigned short* h2b = (unsigned short*)alloc((size_t)N_NODES * HID * 2);
  float* ssrc = (float*)alloc((size_t)N_NODES * 4);
  float* sdst = (float*)alloc((size_t)N_NODES * 4);
  float* rw   = (float*)alloc((size_t)N_EDGES * 4);
  float* acc  = (float*)alloc(4);

  const int t = 256;
  const long long tot = (long long)N_EDGES * HID;
  const unsigned aggBlocks = (unsigned)((tot + t - 1) / t);

  // --- prep: bf16 casts + weight transposes, zero l1 accumulator ---
  k_f32_to_bf16<<<(N_NODES * IN_CH + t - 1) / t, t, 0, stream>>>(x, xb, N_NODES * IN_CH);
  k_transpose_bf16<<<(IN_CH * HID + t - 1) / t, t, 0, stream>>>(Ws1, wts1, IN_CH, HID);
  k_transpose_bf16<<<(IN_CH * HID + t - 1) / t, t, 0, stream>>>(Wn1, wtn1, IN_CH, HID);
  k_transpose_bf16<<<(HID * HID + t - 1) / t, t, 0, stream>>>(Ws2, wts2, HID, HID);
  k_transpose_bf16<<<(HID * HID + t - 1) / t, t, 0, stream>>>(Wn2, wtn2, HID, HID);
  k_transpose_bf16<<<(4 * HID * HID + t - 1) / t, t, 0, stream>>>(Wp1, wtp1, 4 * HID, HID);
  k_zero<<<1, 1, 0, stream>>>(acc);

  // --- stage 1: GCN layer 1 ---
  k_dual_gemm_wmma<<<(N_NODES + 15) / 16, 128, 0, stream>>>(xb, wts1, wtn1, XS, XN,
                                                            N_NODES, IN_CH);
  k_edge_aggregate<<<aggBlocks, t, 0, stream>>>(XN, src, dst, ew, XS, tot);
  k_act_bf16<<<(N_NODES * HID + t - 1) / t, t, 0, stream>>>(XS, b1, h1b, N_NODES);

  // --- stage 2: edge pruning ---
  k_node_scores<<<(N_NODES + t - 1) / t, t, 0, stream>>>(h1b, wsrc, wdst, ssrc, sdst, N_NODES);
  k_edge_probs<<<(N_EDGES + t - 1) / t, t, 0, stream>>>(src, dst, ew, gum, ssrc, sdst,
                                                        be, rw, acc, N_EDGES);

  // --- stage 3: GCN layer 2 on refined graph ---
  k_dual_gemm_wmma<<<(N_NODES + 15) / 16, 128, 0, stream>>>(h1b, wts2, wtn2, XS, XN,
                                                            N_NODES, HID);
  k_edge_aggregate<<<aggBlocks, t, 0, stream>>>(XN, src, dst, rw, XS, tot);
  k_act_bf16<<<(N_NODES * HID + t - 1) / t, t, 0, stream>>>(XS, b2, h2b, N_NODES);

  // --- predictor MLP ---
  k_pred_gemm_wmma<<<(N_PRED + 15) / 16, 128, 0, stream>>>(h1b, h2b, psrc, pdst, wtp1,
                                                           bp1, XN /*hidden*/, N_PRED);
  k_final_logits<<<(N_PRED + t - 1) / t, t, 0, stream>>>(XN, Wp2, bp2, out, N_PRED);
  k_write_l1<<<1, 1, 0, stream>>>(acc, out, N_EDGES, N_PRED);
}
